// HashRelu_88029649698981
// MI455X (gfx1250) — compile-verified, matrix operand order checked by reading, not run on hardware
//
#include <hip/hip_runtime.h>

typedef __attribute__((ext_vector_type(16))) _Float16 v16h;
typedef __attribute__((ext_vector_type(8)))  _Float16 v8h;
typedef __attribute__((ext_vector_type(8)))  float    v8f;

#define HS_MASK ((1u<<20)-1u)
#define WAVES 8
#define THREADS 256

#if defined(__has_builtin)
#if __has_builtin(__builtin_amdgcn_wmma_f16_16x16x32_f16)
#define HAVE_WMMA_F16_ACC 1
#endif
#endif

union Frag32 { v16h v; uint4 q[2]; };
union Frag16 { v8h  v; uint4 q;    };

__device__ __forceinline__ v16h ld_frag2(const _Float16* p0, const _Float16* p1) {
  Frag32 u;
  u.q[0] = *(const uint4*)p0;
  u.q[1] = *(const uint4*)p1;
  return u.v;
}

__device__ __forceinline__ v8h ld_frag1(const _Float16* p0) {
  Frag16 u;
  u.q = *(const uint4*)p0;
  return u.v;
}

// D = A*B + C with f16 accumulate; falls back to f32-acc WMMA if builtin absent.
__device__ __forceinline__ v8h wmma_f16(v16h a, v16h b, v8h c) {
#ifdef HAVE_WMMA_F16_ACC
  return __builtin_amdgcn_wmma_f16_16x16x32_f16(false, a, false, b, (short)0, c, false, false);
#else
  v8f cf;
  #pragma unroll
  for (int i=0;i<8;++i) cf[i] = (float)c[i];
  cf = __builtin_amdgcn_wmma_f32_16x16x32_f16(false, a, false, b, (short)0, cf, false, false);
  v8h d;
  #pragma unroll
  for (int i=0;i<8;++i) d[i] = (_Float16)cf[i];
  return d;
#endif
}

__device__ __forceinline__ v8h relu8(v8h x) {
  const v8h z = {0,0,0,0,0,0,0,0};
  return __builtin_elementwise_max(x, z);   // single v_pk_max_num_f16 per pair
}

// D tiles (f16, N-per-lane) -> next layer's B fragments (pure register repack)
__device__ __forceinline__ void packB(v8h d0, v8h d1, v8h d2, Frag32& c0, Frag32& c1) {
  Frag16 u;
  u.v = d0; c0.q[0] = u.q;
  u.v = d1; c0.q[1] = u.q;
  u.v = d2; c1.q[0] = u.q;
  c1.q[1] = make_uint4(0u,0u,0u,0u);   // K = 48..63 zero padding
}

__global__ __launch_bounds__(THREADS)
void hashmlp_kernel(const float* __restrict__ coords,
                    const float* __restrict__ table,
                    const float* __restrict__ w0, const float* __restrict__ b0,
                    const float* __restrict__ w1, const float* __restrict__ b1,
                    const float* __restrict__ w2, const float* __restrict__ b2,
                    const float* __restrict__ w3, const float* __restrict__ b3,
                    const float* __restrict__ w4, const float* __restrict__ b4,
                    const float* __restrict__ w5, const float* __restrict__ b5,
                    float* __restrict__ out, int nPoints)
{
  // 27 A-fragments (weights) * 512 halves: layer0 -> 3, layers1-4 -> 6 each
  __shared__ __align__(16) _Float16 s_wfrag[27*512];
  __shared__ __align__(16) _Float16 s_biash[5*48];
  __shared__ __align__(16) float    s_w5[48];
  __shared__ float s_b5;
  __shared__ __align__(16) _Float16 s_stage[WAVES][16*32];

  // ---------- one-time per-block: swizzle weights into WMMA A-fragment layout ----------
  {
    const float* Ws[5] = {w0,w1,w2,w3,w4};
    const float* Bs[5] = {b0,b1,b2,b3,b4};
    const int fragBase[5] = {0,3,9,15,21};
    #pragma unroll
    for (int layer=0; layer<5; ++layer) {
      const int Kc  = (layer==0)?1:2;   // K-chunks of 32
      const int Kin = (layer==0)?32:48;
      const int elems = 3*Kc*512;
      const float* W = Ws[layer];
      for (int e=threadIdx.x; e<elems; e+=THREADS) {
        int slot  = e & 15;
        int lane  = (e>>4)&31;
        int rest  = e>>9;
        int chunk = rest % Kc;
        int t     = rest / Kc;
        int n  = t*16 + (lane&15);        // out-feature row (A-matrix M)
        int kbf = (lane>>4)*8;
        int ko = (slot<8) ? (kbf+slot) : (kbf+16+slot-8);
        int k  = chunk*32 + ko;           // in-feature (A-matrix K)
        float v = (k < Kin) ? W[n*Kin + k] : 0.0f;
        s_wfrag[(size_t)(fragBase[layer]+t*Kc+chunk)*512 + lane*16 + slot] = (_Float16)v;
      }
    }
    for (int i=threadIdx.x; i<240; i+=THREADS)
      s_biash[i] = (_Float16)Bs[i/48][i%48];
    for (int i=threadIdx.x; i<48; i+=THREADS)
      s_w5[i] = w5[i];
    if (threadIdx.x==0) s_b5 = b5[0];
  }
  __syncthreads();

  const int wave = threadIdx.x>>5;
  const int l    = threadIdx.x & 31;
  _Float16* stg = s_stage[wave];

  const int nTiles    = nPoints >> 4;
  const int waveCount = gridDim.x * WAVES;
  const int kb0 = (l>>4)*8;
  const int h8  = (l>>4)*8;
  const _Float16* wbase = s_wfrag + l*16;   // per-lane base; frag offsets are immediates

  for (int tile = blockIdx.x*WAVES + wave; tile < nTiles; tile += waveCount) {
    // prefetch next tile's coords while we work on this one
    __builtin_prefetch(coords + (size_t)(tile + waveCount)*32 + l, 0, 1);

    // 16 points * (x,y): one float per lane, broadcast with shuffles
    float cb = coords[(size_t)tile*32 + l];

    // ---------- hash-grid encoding: lanes 0-15 level 2i, lanes 16-31 level 2i+1 ----------
    #pragma unroll
    for (int i=0; i<8; ++i) {
      const int lvl = i*2 + (l>>4);
      const int p   = l & 15;
      float x = __shfl(cb, p*2,   32);
      float y = __shfl(cb, p*2+1, 32);
      const int   res   = 16<<lvl;
      const float scale = (float)res - 1.0f;
      float px = x*scale + 0.5f;
      float py = y*scale + 0.5f;
      float fx = floorf(px), fy = floorf(py);
      float rx = px-fx, ry = py-fy;
      int ix = (int)fx, iy = (int)fy;
      unsigned off;
      const bool dense = (lvl<=6);
      if (dense) off = (((1u<<(2*lvl))-1u)/3u)*256u;          // 256*(4^l-1)/3
      else       off = 1398016u + (unsigned)(lvl-7)*(1u<<20);
      float ax=0.f, ay=0.f;
      #pragma unroll
      for (int dy=0; dy<2; ++dy) {
        #pragma unroll
        for (int dx=0; dx<2; ++dx) {
          int cx = ix+dx, cy = iy+dy;
          unsigned idx;
          if (dense) idx = (unsigned)(cx + cy*res);
          else       idx = ((unsigned)cx ^ ((unsigned)cy*2654435761u)) & HS_MASK;
          const float2 f = *(const float2*)(table + (size_t)(off+idx)*2u);
          float w = (dx? rx : 1.f-rx) * (dy? ry : 1.f-ry);
          ax += w*f.x; ay += w*f.y;
        }
      }
      union { _Float16 h[2]; unsigned u; } pk;
      pk.h[0]=(_Float16)ax; pk.h[1]=(_Float16)ay;
      *(unsigned*)(&stg[p*32 + lvl*2]) = pk.u;   // row = point, col = feature k = 2*lvl
    }
    asm volatile("s_wait_dscnt 0" ::: "memory");

    // B0: encoded features as B operand ([32 feats x 16 points] column-per-lane)
    const _Float16* brow = stg + (l&15)*32;
    v16h bop0 = ld_frag2(brow + kb0, brow + kb0 + 16);

    Frag32 bc0, bc1;

    // ---------- layer 0: D = W0[48x32] * B0 + bias ----------
    {
      v16h a[3]; v8h cf[3];
      #pragma unroll
      for (int t=0; t<3; ++t) {
        const _Float16* ap = wbase + (size_t)t*512;
        a[t]  = ld_frag2(ap, ap+8);
        cf[t] = ld_frag1(s_biash + t*16 + h8);
      }
      v8h d[3];
      #pragma unroll
      for (int t=0; t<3; ++t)
        d[t] = relu8(wmma_f16(a[t], bop0, cf[t]));
      packB(d[0], d[1], d[2], bc0, bc1);
    }

    // ---------- layers 1..3: D = W[48x48] * H + bias (K padded to 64) ----------
    #pragma unroll
    for (int layer=1; layer<4; ++layer) {
      const int base = 3 + (layer-1)*6;
      v16h a[6]; v8h cf[3];
      #pragma unroll
      for (int t=0; t<3; ++t) {
        const _Float16* ap0 = wbase + (size_t)(base + t*2 + 0)*512;
        const _Float16* ap1 = wbase + (size_t)(base + t*2 + 1)*512;
        a[t*2+0] = ld_frag2(ap0, ap0+8);
        a[t*2+1] = ld_frag2(ap1, ap1+8);
        cf[t]    = ld_frag1(s_biash + layer*48 + t*16 + h8);
      }
      v8h d[3];
      #pragma unroll
      for (int t=0; t<3; ++t) {
        v8h c = wmma_f16(a[t*2+0], bc0.v, cf[t]);
        c     = wmma_f16(a[t*2+1], bc1.v, c);
        d[t]  = relu8(c);
      }
      packB(d[0], d[1], d[2], bc0, bc1);
    }

    // ---------- layer 4 + layer 5 fused: relu(W4*H+b4), then dot with w5 ----------
    float acc5 = 0.f;
    {
      const int base = 21;
      v16h a[6]; v8h cf[3];
      #pragma unroll
      for (int t=0; t<3; ++t) {
        const _Float16* ap0 = wbase + (size_t)(base + t*2 + 0)*512;
        const _Float16* ap1 = wbase + (size_t)(base + t*2 + 1)*512;
        a[t*2+0] = ld_frag2(ap0, ap0+8);
        a[t*2+1] = ld_frag2(ap1, ap1+8);
        cf[t]    = ld_frag1(s_biash + 4*48 + t*16 + h8);
      }
      #pragma unroll
      for (int t=0; t<3; ++t) {
        v8h c = wmma_f16(a[t*2+0], bc0.v, cf[t]);
        c     = wmma_f16(a[t*2+1], bc1.v, c);
        c     = relu8(c);
        const float* wv = s_w5 + t*16 + h8;   // lane's features: t*16 + h8 + j
        #pragma unroll
        for (int j=0; j<8; ++j) acc5 += (float)c[j] * wv[j];
      }
    }
    acc5 += __shfl_xor(acc5, 16, 32);
    if (l < 16) out[(size_t)tile*16 + (l&15)] = acc5 + s_b5;
  }
}

extern "C" void kernel_launch(void* const* d_in, const int* in_sizes, int n_in,
                              void* d_out, int out_size, void* d_ws, size_t ws_size,
                              hipStream_t stream) {
  const float* coords = (const float*)d_in[0];
  const float* table  = (const float*)d_in[1];
  const float* w0 = (const float*)d_in[2];  const float* b0 = (const float*)d_in[3];
  const float* w1 = (const float*)d_in[4];  const float* b1 = (const float*)d_in[5];
  const float* w2 = (const float*)d_in[6];  const float* b2 = (const float*)d_in[7];
  const float* w3 = (const float*)d_in[8];  const float* b3 = (const float*)d_in[9];
  const float* w4 = (const float*)d_in[10]; const float* b4 = (const float*)d_in[11];
  const float* w5 = (const float*)d_in[12]; const float* b5 = (const float*)d_in[13];
  float* out = (float*)d_out;

  int nPoints = in_sizes[0] / 2;
  int nTiles  = nPoints / 16;
  int blocks  = (nTiles + WAVES - 1) / WAVES;
  if (blocks > 2048) blocks = 2048;   // ~8 tiles per wave; plenty of waves in flight

  hipLaunchKernelGGL(hashmlp_kernel, dim3(blocks), dim3(THREADS), 0, stream,
                     coords, table, w0, b0, w1, b1, w2, b2, w3, b3, w4, b4, w5, b5,
                     out, nPoints);
  (void)d_ws; (void)ws_size; (void)n_in; (void)out_size;
}
